// POSGatedAttentionLayer_79869211836540
// MI455X (gfx1250) — matrix operand order repaired and checked by reading, not compiled, not run on hardware
//
#include <hip/hip_runtime.h>

// ---------------- problem constants ----------------
#define D_MODEL   1024
#define NHEAD     16
#define HEAD_DIM  64
#define POS_DIM   32
#define POS_HIDDEN 32
#define D_FF      4096
#define BATCH     2
#define SEQ       2048
#define MROWS     (BATCH*SEQ)          // 4096 token rows
#define LN_EPS    1e-5f

// ---------------- WMMA vector types (gfx1250, wave32) ----------------
typedef __attribute__((ext_vector_type(16))) __bf16 v16bf;
typedef __attribute__((ext_vector_type(8)))  float  v8f;

static __device__ __forceinline__ __bf16 f2bf(float f) {
  union { float f; unsigned u; } v; v.f = f;
  unsigned r = v.u + 0x7FFFu + ((v.u >> 16) & 1u);   // round-nearest-even
  unsigned short h = (unsigned short)(r >> 16);
  __bf16 b; __builtin_memcpy(&b, &h, 2); return b;
}
static __device__ __forceinline__ float gelu_exact(float x) {
  return 0.5f * x * (1.0f + erff(x * 0.70710678118654752f));
}

// ---------------- fp32 -> bf16 conversion ----------------
__global__ void cvt_f32_to_bf16(const float* __restrict__ in, __bf16* __restrict__ out, int n) {
  for (int i = blockIdx.x * blockDim.x + threadIdx.x; i < n; i += gridDim.x * blockDim.x)
    out[i] = f2bf(in[i]);
}

// ---------------- POS bias MLP: [B*L,32] -> relu(32) -> [B*L,16] ----------------
__global__ void pos_mlp_kernel(const float* __restrict__ pf,
                               const float* __restrict__ w1, const float* __restrict__ b1,
                               const float* __restrict__ w2, const float* __restrict__ b2,
                               float* __restrict__ out) {
  int row = blockIdx.x * blockDim.x + threadIdx.x;
  if (row >= MROWS) return;
  float x[POS_DIM];
#pragma unroll
  for (int i = 0; i < POS_DIM; i++) x[i] = pf[row * POS_DIM + i];
  float h[POS_HIDDEN];
  for (int j = 0; j < POS_HIDDEN; j++) {
    float s = b1[j];
#pragma unroll
    for (int i = 0; i < POS_DIM; i++) s += x[i] * w1[i * POS_HIDDEN + j];
    h[j] = fmaxf(s, 0.0f);
  }
  for (int o = 0; o < NHEAD; o++) {
    float s = b2[o];
#pragma unroll
    for (int j = 0; j < POS_HIDDEN; j++) s += h[j] * w2[j * NHEAD + o];
    out[row * NHEAD + o] = s;
  }
}

// ---- fragment loaders: contiguous per-lane reads -> ds_load_b128 ----
// A(16x32): lane l holds row l&15; 16 bf16 split as two contiguous 8-element
// chunks at K offsets half*8 and 16+half*8 (CDNA5 ISA 7.12.2).
static __device__ __forceinline__ v16bf load_a_frag(const __bf16* rowp, int ks32, int half) {
  v16bf a;
  __builtin_memcpy(&a,             (const void*)(rowp + ks32 + half * 8), 16);
  __builtin_memcpy((char*)&a + 16, (const void*)(rowp + ks32 + 16 + half * 8), 16);
  return a;
}
// B(32x16): same K split as A: two contiguous 8-element chunks at K offsets
// half*8 and 16+half*8, with B stored K-contiguous per output column.
static __device__ __forceinline__ v16bf load_b_frag(const __bf16* colp, int ks32, int half) {
  v16bf b;
  __builtin_memcpy(&b,             (const void*)(colp + ks32 + half * 8), 16);
  __builtin_memcpy((char*)&b + 16, (const void*)(colp + ks32 + 16 + half * 8), 16);
  return b;
}

// ---------------- generic bf16 WMMA GEMM: C[M,N] = A[M,K] @ W[K,N] + bias ----------------
// Block tile 64x64, K-step 32; DOUBLE-BUFFERED LDS pipeline:
//   iter s: issue global loads for tile s+1 -> regs, WMMA from LDS[s&1],
//           store regs -> LDS[(s+1)&1], one barrier per step.
// B tile staged TRANSPOSED (BsT[n][k]) so B fragments are contiguous reads.
template<int ACT, bool OF32, bool OBF16>
__global__ __launch_bounds__(128) void gemm_bf16(const __bf16* __restrict__ A,
                                                 const __bf16* __restrict__ Bw,
                                                 const float*  __restrict__ bias,
                                                 float*  __restrict__ outF,
                                                 __bf16* __restrict__ outB,
                                                 int Mdim, int Ndim, int Kdim) {
  __shared__ __attribute__((aligned(16))) __bf16 As[2][64][40];    // [buf][m][k]
  __shared__ __attribute__((aligned(16))) __bf16 BsT[2][64][40];   // [buf][n][k]
  const int tid  = threadIdx.x;
  const int lane = tid & 31, wave = tid >> 5;
  const int half = lane >> 4, ln = lane & 15;
  const int bm = blockIdx.x * 64, bn = blockIdx.y * 64;

  // staging indices
  const int akr = tid >> 2, akc = (tid & 3) << 3;   // A: row (0..31, +32), col chunk
  const int bkp = tid >> 3, bnc = (tid & 7) << 3;   // B: k-pair (0..15), n chunk

  int4 ra0, ra1, rb0, rb1;
  auto gload = [&](int k0) {
    ra0 = *reinterpret_cast<const int4*>(A + (size_t)(bm + akr)      * Kdim + k0 + akc);
    ra1 = *reinterpret_cast<const int4*>(A + (size_t)(bm + akr + 32) * Kdim + k0 + akc);
    const __bf16* gb = Bw + (size_t)(k0 + 2 * bkp) * Ndim + bn + bnc;
    rb0 = *reinterpret_cast<const int4*>(gb);
    rb1 = *reinterpret_cast<const int4*>(gb + Ndim);
  };
  auto sstore = [&](int buf) {
    *reinterpret_cast<int4*>(&As[buf][akr][akc])      = ra0;
    *reinterpret_cast<int4*>(&As[buf][akr + 32][akc]) = ra1;
    unsigned short s0[8], s1[8];
    __builtin_memcpy(s0, &rb0, 16);
    __builtin_memcpy(s1, &rb1, 16);
#pragma unroll
    for (int j = 0; j < 8; j++) {   // pack (k,k+1) bf16 pairs -> b32 stores
      unsigned pv = (unsigned)s0[j] | ((unsigned)s1[j] << 16);
      __builtin_memcpy(&BsT[buf][bnc + j][2 * bkp], &pv, 4);
    }
  };

  v8f acc[4] = {};
  const int nsteps = Kdim >> 5;
  gload(0);
  sstore(0);
  __syncthreads();

  for (int s = 0; s < nsteps; s++) {
    const int buf = s & 1;
    if (s + 1 < nsteps) gload((s + 1) << 5);              // overlap HBM with WMMA
    if (s + 2 < nsteps) {                                  // L2 prefetch 2 steps ahead
      __builtin_prefetch(A + (size_t)(bm + akr) * Kdim + ((s + 2) << 5) + akc, 0, 1);
      __builtin_prefetch(Bw + (size_t)(((s + 2) << 5) + 2 * bkp) * Ndim + bn + bnc, 0, 1);
    }

    const v16bf a = load_a_frag(&As[buf][wave * 16 + ln][0], 0, half);
#pragma unroll
    for (int t = 0; t < 4; t++) {
      const v16bf bb = load_b_frag(&BsT[buf][t * 16 + ln][0], 0, half);
      acc[t] = __builtin_amdgcn_wmma_f32_16x16x32_bf16(false, a, false, bb,
                                                       (short)0, acc[t], false, false);
    }

    if (s + 1 < nsteps) {
      sstore(buf ^ 1);     // fill the other buffer (nobody reads it this step)
      __syncthreads();     // one barrier per K-step
    }
  }

#pragma unroll
  for (int t = 0; t < 4; t++) {
    int col = bn + t * 16 + ln;
    float bv = bias ? bias[col] : 0.0f;
#pragma unroll
    for (int v = 0; v < 8; v++) {
      int row = bm + wave * 16 + v + half * 8;
      float x = acc[t][v] + bv;
      if (ACT == 1) x = gelu_exact(x);
      size_t o = (size_t)row * Ndim + col;
      if (OF32)  outF[o] = x;
      if (OBF16) outB[o] = f2bf(x);
    }
  }
}

// ---------------- flash attention + entropy ----------------
// grid: (SEQ/64 query tiles, BATCH*NHEAD); 128 threads = 4 waves.
// S = Q K^T (WMMA), online softmax with per-row stats (m, Z, T=sum (s-m)e^{s-m});
// entropy = log Z - T/Z. O += P V with V staged transposed (VsT[d][key]).
__global__ __launch_bounds__(128) void attn_kernel(const __bf16* __restrict__ Qb,
                                                   const __bf16* __restrict__ Kb,
                                                   const __bf16* __restrict__ Vb,
                                                   const float* __restrict__ qbias,
                                                   const float* __restrict__ kbias,
                                                   const float* __restrict__ temp,
                                                   __bf16* __restrict__ Ob,
                                                   float* __restrict__ entAcc) {
  __shared__ __attribute__((aligned(16))) __bf16 Qs[64][72];    // [q][d]
  __shared__ __attribute__((aligned(16))) __bf16 Ks[64][72];    // [key][d]
  __shared__ __attribute__((aligned(16))) __bf16 VsT[64][72];   // [d][key]
  __shared__ __attribute__((aligned(16))) __bf16 Ps[64][72];    // [q][key]
  __shared__ float kb_s[64], qb_s[64];
  __shared__ float mrow[64], Zrow[64], Trow[64], srow[64];

  const int tid  = threadIdx.x;
  const int lane = tid & 31, wave = tid >> 5;
  const int half = lane >> 4, ln = lane & 15;
  const int b = blockIdx.y / NHEAD, h = blockIdx.y % NHEAD;
  const int l0 = blockIdx.x * 64;

  float tc = fminf(fmaxf(temp[h], 0.1f), 5.0f);
  const float invt = 1.0f / tc;
  const float sfac = 0.125f * invt;   // 1/sqrt(64) / temp

  // stage Q tile (64x64 bf16)
#pragma unroll
  for (int i = 0; i < 4; i++) {
    int e = tid + i * 128;
    int r = e >> 3, c = (e & 7) << 3;
    *reinterpret_cast<int4*>(&Qs[r][c]) = *reinterpret_cast<const int4*>(
        Qb + (size_t)(b * SEQ + l0 + r) * D_MODEL + h * HEAD_DIM + c);
  }
  if (tid < 64) {
    qb_s[tid] = qbias[(size_t)(b * SEQ + l0 + tid) * NHEAD + h] * invt;
    mrow[tid] = -1e30f; Zrow[tid] = 0.0f; Trow[tid] = 0.0f;
  }
  v8f oacc[4] = {};
  __syncthreads();

  for (int k0 = 0; k0 < SEQ; k0 += 64) {
    // prefetch next key tile
    if (k0 + 64 < SEQ) {
      size_t gp = (size_t)(b * SEQ + k0 + 64 + (tid >> 3)) * D_MODEL + h * HEAD_DIM + ((tid & 7) << 3);
      __builtin_prefetch(Kb + gp, 0, 1);
      __builtin_prefetch(Vb + gp, 0, 1);
    }
    // stage K tile row-major (64x64)
#pragma unroll
    for (int i = 0; i < 4; i++) {
      int e = tid + i * 128;
      int r = e >> 3, c = (e & 7) << 3;
      *reinterpret_cast<int4*>(&Ks[r][c]) = *reinterpret_cast<const int4*>(
          Kb + (size_t)(b * SEQ + k0 + r) * D_MODEL + h * HEAD_DIM + c);
    }
    // stage V tile transposed: VsT[d][key]; thread handles key pair + d chunk
#pragma unroll
    for (int u = 0; u < 2; u++) {
      int unit = tid + u * 128;          // 0..255
      int kp = unit >> 3;                // key pair (0..31)
      int nc = (unit & 7) << 3;          // d chunk
      const __bf16* gv = Vb + (size_t)(b * SEQ + k0 + 2 * kp) * D_MODEL + h * HEAD_DIM + nc;
      int4 r0 = *reinterpret_cast<const int4*>(gv);
      int4 r1 = *reinterpret_cast<const int4*>(gv + D_MODEL);
      unsigned short s0[8], s1[8];
      __builtin_memcpy(s0, &r0, 16);
      __builtin_memcpy(s1, &r1, 16);
#pragma unroll
      for (int j = 0; j < 8; j++) {
        unsigned pv = (unsigned)s0[j] | ((unsigned)s1[j] << 16);
        __builtin_memcpy(&VsT[nc + j][2 * kp], &pv, 4);
      }
    }
    if (tid < 64) kb_s[tid] = kbias[(size_t)(b * SEQ + k0 + tid) * NHEAD + h] * invt;
    __syncthreads();

    // S = Q K^T (contraction over head dim, 2 x K=32 steps)
    v8f sacc[4] = {};
#pragma unroll
    for (int ks = 0; ks < 2; ks++) {
      const v16bf a = load_a_frag(&Qs[wave * 16 + ln][0], ks * 32, half);
#pragma unroll
      for (int t = 0; t < 4; t++) {
        const v16bf bb = load_b_frag(&Ks[t * 16 + ln][0], ks * 32, half);  // B[k=d][n=key]
        sacc[t] = __builtin_amdgcn_wmma_f32_16x16x32_bf16(false, a, false, bb,
                                                          (short)0, sacc[t], false, false);
      }
    }

    // scale + biases, row max
    float sv[4][8], rmax[8];
#pragma unroll
    for (int v = 0; v < 8; v++) rmax[v] = -1e30f;
#pragma unroll
    for (int t = 0; t < 4; t++) {
      float kbv = kb_s[t * 16 + ln];
#pragma unroll
      for (int v = 0; v < 8; v++) {
        float s = sacc[t][v] * sfac + qb_s[wave * 16 + v + half * 8] + kbv;
        sv[t][v] = s;
        rmax[v] = fmaxf(rmax[v], s);
      }
    }
#pragma unroll
    for (int v = 0; v < 8; v++) {
#pragma unroll
      for (int m = 8; m >= 1; m >>= 1) rmax[v] = fmaxf(rmax[v], __shfl_xor(rmax[v], m, 32));
    }
    float newm[8], rsum[8], rent[8];
#pragma unroll
    for (int v = 0; v < 8; v++) {
      newm[v] = fmaxf(mrow[wave * 16 + v + half * 8], rmax[v]);
      rsum[v] = 0.0f; rent[v] = 0.0f;
    }
#pragma unroll
    for (int t = 0; t < 4; t++) {
#pragma unroll
      for (int v = 0; v < 8; v++) {
        float sp = sv[t][v] - newm[v];
        float p = __expf(sp);
        rsum[v] += p;
        rent[v] += sp * p;
        Ps[wave * 16 + v + half * 8][t * 16 + ln] = f2bf(p);
      }
    }
#pragma unroll
    for (int v = 0; v < 8; v++) {
#pragma unroll
      for (int m = 8; m >= 1; m >>= 1) {
        rsum[v] += __shfl_xor(rsum[v], m, 32);
        rent[v] += __shfl_xor(rent[v], m, 32);
      }
    }
    if (ln == 0) {   // one lane per half updates its 8 rows
#pragma unroll
      for (int v = 0; v < 8; v++) {
        int r = wave * 16 + v + half * 8;
        float om = mrow[r], nm = newm[v];
        float d = om - nm;
        float esc = __expf(d);
        float zo = Zrow[r];
        Trow[r] = esc * (Trow[r] + d * zo) + rent[v];
        Zrow[r] = esc * zo + rsum[v];
        mrow[r] = nm;
        srow[r] = esc;
      }
    }
    __syncthreads();

    // rescale O, then O += P V
#pragma unroll
    for (int t = 0; t < 4; t++) {
#pragma unroll
      for (int v = 0; v < 8; v++) oacc[t][v] *= srow[wave * 16 + v + half * 8];
    }
#pragma unroll
    for (int ks = 0; ks < 2; ks++) {
      const v16bf a = load_a_frag(&Ps[wave * 16 + ln][0], ks * 32, half);
#pragma unroll
      for (int t = 0; t < 4; t++) {
        const v16bf bb = load_b_frag(&VsT[t * 16 + ln][0], ks * 32, half); // B[k=key][n=d]
        oacc[t] = __builtin_amdgcn_wmma_f32_16x16x32_bf16(false, a, false, bb,
                                                          (short)0, oacc[t], false, false);
      }
    }
    __syncthreads();
  }

  // finalize: O /= Z, write bf16; entropy per row
#pragma unroll
  for (int t = 0; t < 4; t++) {
#pragma unroll
    for (int v = 0; v < 8; v++) {
      int r = wave * 16 + v + half * 8;
      float o = oacc[t][v] / Zrow[r];
      Ob[(size_t)(b * SEQ + l0 + r) * D_MODEL + h * HEAD_DIM + t * 16 + ln] = f2bf(o);
    }
  }
  if (tid < 64) {
    float z = Zrow[tid];
    atomicAdd(entAcc, logf(z) - Trow[tid] / z);
  }
}

// ---------------- residual + LayerNorm (one block per token row) ----------------
template<bool OBF16>
__global__ __launch_bounds__(256) void resid_ln(const float* __restrict__ X,
                                                const float* __restrict__ Y,
                                                const float* __restrict__ g,
                                                const float* __restrict__ bb,
                                                float* __restrict__ outF,
                                                __bf16* __restrict__ outB) {
  __shared__ float red[256];
  const int row = blockIdx.x, tid = threadIdx.x;
  const size_t base = (size_t)row * D_MODEL;
  float v[4]; float s = 0.0f;
#pragma unroll
  for (int i = 0; i < 4; i++) {
    int c = tid + i * 256;
    v[i] = X[base + c] + Y[base + c];
    s += v[i];
  }
  red[tid] = s; __syncthreads();
  for (int o = 128; o > 0; o >>= 1) { if (tid < o) red[tid] += red[tid + o]; __syncthreads(); }
  float mu = red[0] * (1.0f / D_MODEL);
  __syncthreads();
  float q = 0.0f;
#pragma unroll
  for (int i = 0; i < 4; i++) { float d = v[i] - mu; q += d * d; }
  red[tid] = q; __syncthreads();
  for (int o = 128; o > 0; o >>= 1) { if (tid < o) red[tid] += red[tid + o]; __syncthreads(); }
  float rstd = rsqrtf(red[0] * (1.0f / D_MODEL) + LN_EPS);
#pragma unroll
  for (int i = 0; i < 4; i++) {
    int c = tid + i * 256;
    float o = (v[i] - mu) * rstd * g[c] + bb[c];
    outF[base + c] = o;
    if (OBF16) outB[base + c] = f2bf(o);
  }
}

__global__ void zero1(float* p) { if (threadIdx.x == 0 && blockIdx.x == 0) p[0] = 0.0f; }
__global__ void ent_final(const float* __restrict__ ent, float* __restrict__ out) {
  out[0] = ent[0] * (1.0f / (float)(BATCH * NHEAD * SEQ));
}

// ---------------- host orchestration ----------------
extern "C" void kernel_launch(void* const* d_in, const int* in_sizes, int n_in,
                              void* d_out, int out_size, void* d_ws, size_t ws_size,
                              hipStream_t stream) {
  (void)in_sizes; (void)n_in; (void)out_size; (void)ws_size;
  const float* src  = (const float*)d_in[0];
  const float* pos  = (const float*)d_in[1];
  const float* Wq   = (const float*)d_in[2];  const float* bq   = (const float*)d_in[3];
  const float* Wk   = (const float*)d_in[4];  const float* bk   = (const float*)d_in[5];
  const float* Wv   = (const float*)d_in[6];  const float* bv   = (const float*)d_in[7];
  const float* Wo   = (const float*)d_in[8];  const float* bo   = (const float*)d_in[9];
  const float* pqw1 = (const float*)d_in[10]; const float* pqb1 = (const float*)d_in[11];
  const float* pqw2 = (const float*)d_in[12]; const float* pqb2 = (const float*)d_in[13];
  const float* pkw1 = (const float*)d_in[14]; const float* pkb1 = (const float*)d_in[15];
  const float* pkw2 = (const float*)d_in[16]; const float* pkb2 = (const float*)d_in[17];
  const float* temp = (const float*)d_in[18];
  const float* ln1g = (const float*)d_in[19]; const float* ln1b = (const float*)d_in[20];
  const float* ln2g = (const float*)d_in[21]; const float* ln2b = (const float*)d_in[22];
  const float* W1   = (const float*)d_in[23]; const float* b1   = (const float*)d_in[24];
  const float* W2   = (const float*)d_in[25]; const float* b2   = (const float*)d_in[26];
  float* out = (float*)d_out;

  char* ws = (char*)d_ws;
  size_t off = 0;
  auto carve = [&](size_t bytes) -> char* {
    char* p = ws + off;
    off += (bytes + 255) & ~(size_t)255;
    return p;
  };
  const size_t MD = (size_t)MROWS * D_MODEL;
  __bf16* srcb = (__bf16*)carve(MD * 2);
  __bf16* wqb  = (__bf16*)carve((size_t)D_MODEL * D_MODEL * 2);
  __bf16* wkb  = (__bf16*)carve((size_t)D_MODEL * D_MODEL * 2);
  __bf16* wvb  = (__bf16*)carve((size_t)D_MODEL * D_MODEL * 2);
  __bf16* wob  = (__bf16*)carve((size_t)D_MODEL * D_MODEL * 2);
  __bf16* w1b  = (__bf16*)carve((size_t)D_MODEL * D_FF * 2);
  __bf16* w2b  = (__bf16*)carve((size_t)D_FF * D_MODEL * 2);
  __bf16* Qbq  = (__bf16*)carve(MD * 2);
  __bf16* Kbq  = (__bf16*)carve(MD * 2);
  __bf16* Vbq  = (__bf16*)carve(MD * 2);
  __bf16* attb = (__bf16*)carve(MD * 2);
  __bf16* xb   = (__bf16*)carve(MD * 2);
  __bf16* ff1b = (__bf16*)carve((size_t)MROWS * D_FF * 2);
  float*  qbias = (float*)carve((size_t)MROWS * NHEAD * 4);
  float*  kbias = (float*)carve((size_t)MROWS * NHEAD * 4);
  float*  projf = (float*)carve(MD * 4);
  float*  xf    = (float*)carve(MD * 4);
  float*  ff2f  = (float*)carve(MD * 4);
  float*  ent   = (float*)carve(256);

  // fp32 -> bf16 staging
  cvt_f32_to_bf16<<<1024, 256, 0, stream>>>(src, srcb, (int)MD);
  cvt_f32_to_bf16<<<512,  256, 0, stream>>>(Wq, wqb, D_MODEL * D_MODEL);
  cvt_f32_to_bf16<<<512,  256, 0, stream>>>(Wk, wkb, D_MODEL * D_MODEL);
  cvt_f32_to_bf16<<<512,  256, 0, stream>>>(Wv, wvb, D_MODEL * D_MODEL);
  cvt_f32_to_bf16<<<512,  256, 0, stream>>>(Wo, wob, D_MODEL * D_MODEL);
  cvt_f32_to_bf16<<<1024, 256, 0, stream>>>(W1, w1b, D_MODEL * D_FF);
  cvt_f32_to_bf16<<<1024, 256, 0, stream>>>(W2, w2b, D_FF * D_MODEL);

  // POS bias adapters
  pos_mlp_kernel<<<MROWS / 256, 256, 0, stream>>>(pos, pqw1, pqb1, pqw2, pqb2, qbias);
  pos_mlp_kernel<<<MROWS / 256, 256, 0, stream>>>(pos, pkw1, pkb1, pkw2, pkb2, kbias);

  zero1<<<1, 1, 0, stream>>>(ent);

  dim3 blk(128);
  dim3 gDD(MROWS / 64, D_MODEL / 64);
  // QKV projections -> bf16
  gemm_bf16<0, false, true><<<gDD, blk, 0, stream>>>(srcb, wqb, bq, nullptr, Qbq, MROWS, D_MODEL, D_MODEL);
  gemm_bf16<0, false, true><<<gDD, blk, 0, stream>>>(srcb, wkb, bk, nullptr, Kbq, MROWS, D_MODEL, D_MODEL);
  gemm_bf16<0, false, true><<<gDD, blk, 0, stream>>>(srcb, wvb, bv, nullptr, Vbq, MROWS, D_MODEL, D_MODEL);

  // flash attention (+ entropy accumulation)
  attn_kernel<<<dim3(SEQ / 64, BATCH * NHEAD), blk, 0, stream>>>(Qbq, Kbq, Vbq, qbias, kbias,
                                                                 temp, attb, ent);

  // output projection -> f32
  gemm_bf16<0, true, false><<<gDD, blk, 0, stream>>>(attb, wob, bo, projf, nullptr, MROWS, D_MODEL, D_MODEL);

  // x = LN1(src + attn_out) -> f32 + bf16
  resid_ln<true><<<MROWS, 256, 0, stream>>>(src, projf, ln1g, ln1b, xf, xb);

  // FFN: GELU(x W1 + b1) W2 + b2
  gemm_bf16<1, false, true><<<dim3(MROWS / 64, D_FF / 64), blk, 0, stream>>>(xb, w1b, b1, nullptr, ff1b,
                                                                             MROWS, D_FF, D_MODEL);
  gemm_bf16<0, true, false><<<gDD, blk, 0, stream>>>(ff1b, w2b, b2, ff2f, nullptr, MROWS, D_MODEL, D_FF);

  // out = LN2(x + ff)
  resid_ln<false><<<MROWS, 256, 0, stream>>>(xf, ff2f, ln2g, ln2b, out, nullptr);

  // entropy mean scalar appended after the [B,L,D] output
  ent_final<<<1, 1, 0, stream>>>(ent, out + MD);
}
